// LSTM_Gal_21474836480118
// MI455X (gfx1250) — compile-verified, matrix operand order checked by reading, not run on hardware
//
#include <hip/hip_runtime.h>

// ---------------------------------------------------------------------------
// Gal-dropout LSTM for MI455X (gfx1250, wave32, WMMA bf16 @ f32 accumulate)
//   B=128, T=512, I=256, H=512
// ---------------------------------------------------------------------------

#define B_ 128
#define T_ 512
#define I_ 256
#define H_ 512

typedef __attribute__((ext_vector_type(16))) __bf16 v16bf;
typedef __attribute__((ext_vector_type(8)))  float  v8f;
typedef __attribute__((ext_vector_type(4)))  float  f32x4;

#define WMMA_BF16(a, b, c) \
    __builtin_amdgcn_wmma_f32_16x16x32_bf16(false, (a), false, (b), (short)0, (c), false, false)

__device__ __forceinline__ float sigmoidf_(float x) {
    return 1.0f / (1.0f + __expf(-x));
}

// A-fragment (16x32 bf16) built from fp32 data * fp32 mask, per-lane:
// lane holds row m, 16 consecutive K values starting at kBase (kBase = kIter + (lane>>4)*16)
__device__ __forceinline__ v16bf load_a_masked(const float* __restrict__ xrow,
                                               const float* __restrict__ mrow,
                                               int kBase) {
    v16bf a;
    const f32x4* xv = (const f32x4*)(xrow + kBase);
    const f32x4* mv = (const f32x4*)(mrow + kBase);
#pragma unroll
    for (int i = 0; i < 4; ++i) {
        f32x4 xx = xv[i];
        f32x4 mm = mv[i];
#pragma unroll
        for (int j = 0; j < 4; ++j)
            a[4 * i + j] = (__bf16)(xx[j] * mm[j]);
    }
    return a;
}

// ---------------------------------------------------------------------------
// Prep: convert 4 fp32 weight matrices to bf16, gate-major
// ---------------------------------------------------------------------------
__global__ void cvt4_kernel(const float* __restrict__ s0, const float* __restrict__ s1,
                            const float* __restrict__ s2, const float* __restrict__ s3,
                            __bf16* __restrict__ dst, int per) {
    int i = blockIdx.x * blockDim.x + threadIdx.x;
    if (i >= 4 * per) return;
    int g = i / per, r = i - g * per;
    const float* s = (g == 0) ? s0 : (g == 1) ? s1 : (g == 2) ? s2 : s3;
    dst[i] = (__bf16)s[r];
}

// zero initial masked-h (bf16) and cell state
__global__ void init_kernel(__bf16* __restrict__ hm0, float* __restrict__ c,
                            int nbf, int nf) {
    int i = blockIdx.x * blockDim.x + threadIdx.x;
    if (i < nbf) hm0[i] = (__bf16)0.0f;
    if (i < nf)  c[i] = 0.0f;
}

// ---------------------------------------------------------------------------
// Input projections: gx[t][g][b][h] = (x[b][t][:]*mask_x[g][b][:]) . W_g[h][:] + bias_g[h]
// One wave per 32x32 tile of the (M = B*T) x (N = H) output, K = I.
// ---------------------------------------------------------------------------
__global__ __launch_bounds__(32)
void input_gemm_kernel(const float* __restrict__ x, const float* __restrict__ mask_x,
                       const __bf16* __restrict__ Wbf,
                       const float* __restrict__ bi, const float* __restrict__ bf2,
                       const float* __restrict__ bc, const float* __restrict__ bo,
                       float* __restrict__ gx) {
    const int lane = (int)threadIdx.x & 31;
    const int half = lane >> 4;
    const int l    = lane & 15;
    const int g    = (int)blockIdx.z;
    const int m0   = (int)blockIdx.x * 32;   // over B*T
    const int n0   = (int)blockIdx.y * 32;   // over H

    const float* biasP = (g == 0) ? bi : (g == 1) ? bf2 : (g == 2) ? bc : bo;

    const float*  xrow[2];
    const float*  mrow[2];
    const __bf16* wrow[2];
#pragma unroll
    for (int i = 0; i < 2; ++i) {
        int m = m0 + i * 16 + l;            // row in B*T
        int b = m >> 9;                     // m / T
        xrow[i] = x + (size_t)m * I_;
        mrow[i] = mask_x + ((size_t)g * B_ + b) * I_;
        int n = n0 + i * 16 + l;
        wrow[i] = Wbf + ((size_t)g * H_ + n) * I_;
    }

    v8f acc[2][2] = {};
    for (int k = 0; k < I_; k += 32) {
        int kb = k + half * 16;
        v16bf a0 = load_a_masked(xrow[0], mrow[0], kb);
        v16bf a1 = load_a_masked(xrow[1], mrow[1], kb);
        v16bf b0 = *(const v16bf*)(wrow[0] + kb);
        v16bf b1 = *(const v16bf*)(wrow[1] + kb);
        acc[0][0] = WMMA_BF16(a0, b0, acc[0][0]);
        acc[0][1] = WMMA_BF16(a0, b1, acc[0][1]);
        acc[1][0] = WMMA_BF16(a1, b0, acc[1][0]);
        acc[1][1] = WMMA_BF16(a1, b1, acc[1][1]);
    }

#pragma unroll
    for (int mi = 0; mi < 2; ++mi)
#pragma unroll
        for (int ni = 0; ni < 2; ++ni) {
            int n = n0 + ni * 16 + l;
            float bias = biasP[n];
#pragma unroll
            for (int v = 0; v < 8; ++v) {
                int m = m0 + mi * 16 + half * 8 + v;
                int b = m >> 9;
                int t = m & (T_ - 1);
                gx[(((size_t)t * 4 + g) * B_ + b) * H_ + n] = acc[mi][ni][v] + bias;
            }
        }
}

// ---------------------------------------------------------------------------
// One recurrence step. Each wave owns a 32x32 (b,h) output tile and computes
// all 4 gate GEMMs (K = H) into registers, then does the LSTM pointwise
// update in registers and emits next step's bf16 masked-h A-matrix.
// ---------------------------------------------------------------------------
__global__ __launch_bounds__(32)
void lstm_step_kernel(const float* __restrict__ gx_t,    // [4][B][H]
                      const __bf16* __restrict__ hm_in,  // [4][B][H]
                      const __bf16* __restrict__ Whbf,   // [4][H][H]
                      const float* __restrict__ mask_h,  // [4][B][H]
                      float* __restrict__ cbuf,          // [B][H]
                      float* __restrict__ hbuf,          // [B][H]
                      __bf16* __restrict__ hm_out,       // [4][B][H]
                      float* __restrict__ ys,            // [B][T][H]
                      int t) {
    const int lane = (int)threadIdx.x & 31;
    const int half = lane >> 4;
    const int l    = lane & 15;
    const int m0   = (int)blockIdx.x * 32;   // over B
    const int n0   = (int)blockIdx.y * 32;   // over H

    v8f acc[4][2][2] = {};

#pragma unroll
    for (int g = 0; g < 4; ++g) {
        const __bf16* arow0 = hm_in + ((size_t)g * B_ + (m0 + l)) * H_;
        const __bf16* arow1 = hm_in + ((size_t)g * B_ + (m0 + 16 + l)) * H_;
        const __bf16* brow0 = Whbf  + ((size_t)g * H_ + (n0 + l)) * H_;
        const __bf16* brow1 = Whbf  + ((size_t)g * H_ + (n0 + 16 + l)) * H_;
        for (int k = 0; k < H_; k += 32) {
            int kb = k + half * 16;
            v16bf a0 = *(const v16bf*)(arow0 + kb);
            v16bf a1 = *(const v16bf*)(arow1 + kb);
            v16bf b0 = *(const v16bf*)(brow0 + kb);
            v16bf b1 = *(const v16bf*)(brow1 + kb);
            acc[g][0][0] = WMMA_BF16(a0, b0, acc[g][0][0]);
            acc[g][0][1] = WMMA_BF16(a0, b1, acc[g][0][1]);
            acc[g][1][0] = WMMA_BF16(a1, b0, acc[g][1][0]);
            acc[g][1][1] = WMMA_BF16(a1, b1, acc[g][1][1]);
        }
    }

    // Pointwise LSTM update — all 4 gate accumulators share the same (m,n)
    // lane mapping, so everything stays in registers.
#pragma unroll
    for (int mi = 0; mi < 2; ++mi)
#pragma unroll
        for (int ni = 0; ni < 2; ++ni)
#pragma unroll
            for (int v = 0; v < 8; ++v) {
                int bidx = m0 + mi * 16 + half * 8 + v;
                int h    = n0 + ni * 16 + l;
                size_t bh = (size_t)bidx * H_ + h;

                float gi = gx_t[((size_t)0 * B_ + bidx) * H_ + h] + acc[0][mi][ni][v];
                float gf = gx_t[((size_t)1 * B_ + bidx) * H_ + h] + acc[1][mi][ni][v];
                float gc = gx_t[((size_t)2 * B_ + bidx) * H_ + h] + acc[2][mi][ni][v];
                float go = gx_t[((size_t)3 * B_ + bidx) * H_ + h] + acc[3][mi][ni][v];

                float it = sigmoidf_(gi);
                float ft = sigmoidf_(gf);
                float ct = tanhf(gc);
                float ot = sigmoidf_(go);

                float cold = cbuf[bh];
                float cn = ft * cold + it * ct;
                float hn = ot * tanhf(cn);

                cbuf[bh] = cn;
                hbuf[bh] = hn;
                ys[((size_t)bidx * T_ + t) * H_ + h] = hn;

#pragma unroll
                for (int g = 0; g < 4; ++g) {
                    size_t gbh = ((size_t)g * B_ + bidx) * H_ + h;
                    hm_out[gbh] = (__bf16)(hn * mask_h[gbh]);
                }
            }
}

__global__ void finalize_kernel(const float* __restrict__ h, const float* __restrict__ c,
                                float* __restrict__ outH, float* __restrict__ outC, int n) {
    int i = blockIdx.x * blockDim.x + threadIdx.x;
    if (i < n) { outH[i] = h[i]; outC[i] = c[i]; }
}

// ---------------------------------------------------------------------------
extern "C" void kernel_launch(void* const* d_in, const int* in_sizes, int n_in,
                              void* d_out, int out_size, void* d_ws, size_t ws_size,
                              hipStream_t stream) {
    const float* x      = (const float*)d_in[0];
    const float* mask_x = (const float*)d_in[1];
    const float* mask_h = (const float*)d_in[2];
    const float* W_ii = (const float*)d_in[3];  const float* b_ii = (const float*)d_in[4];
    const float* W_if = (const float*)d_in[5];  const float* b_if = (const float*)d_in[6];
    const float* W_ic = (const float*)d_in[7];  const float* b_ic = (const float*)d_in[8];
    const float* W_io = (const float*)d_in[9];  const float* b_io = (const float*)d_in[10];
    const float* Wh_i = (const float*)d_in[11];
    const float* Wh_f = (const float*)d_in[12];
    const float* Wh_c = (const float*)d_in[13];
    const float* Wh_o = (const float*)d_in[14];

    float* ys   = (float*)d_out;
    float* outH = ys + (size_t)B_ * T_ * H_;
    float* outC = outH + (size_t)B_ * H_;

    // workspace carve (256B aligned)
    char* p = (char*)d_ws;
    auto carve = [&](size_t bytes) {
        char* r = p;
        p += (bytes + 255) & ~(size_t)255;
        return r;
    };
    __bf16* Wbf  = (__bf16*)carve((size_t)4 * H_ * I_ * sizeof(__bf16)); // 1 MB
    __bf16* Whbf = (__bf16*)carve((size_t)4 * H_ * H_ * sizeof(__bf16)); // 2 MB
    __bf16* hm0  = (__bf16*)carve((size_t)4 * B_ * H_ * sizeof(__bf16)); // 512 KB
    __bf16* hm1  = (__bf16*)carve((size_t)4 * B_ * H_ * sizeof(__bf16)); // 512 KB
    float*  cbuf = (float*)carve((size_t)B_ * H_ * sizeof(float));       // 256 KB
    float*  hbuf = (float*)carve((size_t)B_ * H_ * sizeof(float));       // 256 KB
    float*  gx   = (float*)carve((size_t)T_ * 4 * B_ * H_ * sizeof(float)); // 512 MB

    // 1) weights -> bf16 (gate-major, K contiguous => direct B-fragment loads)
    {
        int per = H_ * I_, n = 4 * per;
        cvt4_kernel<<<(n + 255) / 256, 256, 0, stream>>>(W_ii, W_if, W_ic, W_io, Wbf, per);
    }
    {
        int per = H_ * H_, n = 4 * per;
        cvt4_kernel<<<(n + 255) / 256, 256, 0, stream>>>(Wh_i, Wh_f, Wh_c, Wh_o, Whbf, per);
    }

    // 2) zero h0-derived A-matrix and c0
    {
        int nbf = 4 * B_ * H_, nf = B_ * H_;
        init_kernel<<<(nbf + 255) / 256, 256, 0, stream>>>(hm0, cbuf, nbf, nf);
    }

    // 3) hoisted input projections: 4 x (65536 x 256 x 512) bf16 WMMA GEMM
    {
        dim3 grid((B_ * T_) / 32, H_ / 32, 4);
        input_gemm_kernel<<<grid, 32, 0, stream>>>(x, mask_x, Wbf,
                                                   b_ii, b_if, b_ic, b_io, gx);
    }

    // 4) recurrence: 512 step kernels (stream order provides the dependency),
    //    double-buffered masked-h bf16 A-matrix
    {
        const __bf16* hin = hm0;
        __bf16* hout = hm1;
        dim3 grid(B_ / 32, H_ / 32);
        for (int t = 0; t < T_; ++t) {
            lstm_step_kernel<<<grid, 32, 0, stream>>>(
                gx + (size_t)t * 4 * B_ * H_, hin, Whbf, mask_h,
                cbuf, hbuf, hout, ys, t);
            __bf16* tmp = (__bf16*)hin;
            hin = hout;
            hout = tmp;
        }
    }

    // 5) emit hT, cT
    {
        int n = B_ * H_;
        finalize_kernel<<<(n + 255) / 256, 256, 0, stream>>>(hbuf, cbuf, outH, outC, n);
    }
}
